// _netD_8_66872640799172
// MI455X (gfx1250) — matrix operand order repaired and checked by reading, not compiled, not run on hardware
//
#include <hip/hip_runtime.h>
#include <hip/hip_bf16.h>

typedef __attribute__((ext_vector_type(16))) _Float16 v16h;
typedef __attribute__((ext_vector_type(8)))  _Float16 v8h;
typedef __attribute__((ext_vector_type(8)))  float    v8f;

#define BN_EPS 1e-5f
#define LSLOPE 0.2f

// ---------------------------------------------------------------------------
// CDNA5 async DMA: global -> LDS, 16 bytes, tracked by ASYNCcnt.
// Inline asm (portable across ROCm7.2 / amdgpu-toolchain builtin arity skew).
// ---------------------------------------------------------------------------
__device__ __forceinline__ void async_b128_to_lds(const _Float16* gsrc,
                                                  _Float16* lds_dst) {
  uint32_t loff = (uint32_t)(size_t)lds_dst;  // low 32 bits = LDS byte offset
  asm volatile("global_load_async_to_lds_b128 %0, %1, off"
               :: "v"(loff), "v"(gsrc)
               : "memory");
}

__device__ __forceinline__ void wait_async0() {
  asm volatile("s_wait_asynccnt 0x0" ::: "memory");
}

// ---------------------------------------------------------------------------
// Convert f32 weights (Co x Kc row-major, Kc = Ci*KH*KW) to f16 padded
// (Mpad64 x Kpad) with zero fill (Mpad64 = round-up-to-64 so 4-wave blocks
// always have uniform work).
// ---------------------------------------------------------------------------
__global__ void k_wcvt(const float* __restrict__ w, _Float16* __restrict__ wp,
                       int Co, int Kc, int Kpad, int Mpad) {
  int idx = blockIdx.x * blockDim.x + threadIdx.x;
  int total = Mpad * Kpad;
  if (idx >= total) return;
  int co = idx / Kpad;
  int k  = idx % Kpad;
  float v = (co < Co && k < Kc) ? w[(size_t)co * Kc + k] : 0.0f;
  wp[idx] = (_Float16)v;
}

// ---------------------------------------------------------------------------
// im2col into f16, column-major K-vectors: dst[col*Kpad + k].
// src indexed as src[ci*strideC + n*strideN + h*W + w] (handles NCHW input
// and channel-major intermediates via the two strides).
// ---------------------------------------------------------------------------
__global__ void k_im2col(const float* __restrict__ src, _Float16* __restrict__ dst,
                         int H, int W, int Ho, int Wo,
                         int KH, int KW, int stride, int pad,
                         long strideC, long strideN,
                         int Kc, int Kpad, long total) {
  long idx = (long)blockIdx.x * blockDim.x + threadIdx.x;
  if (idx >= total) return;
  int  k   = (int)(idx % Kpad);
  long col = idx / Kpad;
  float v = 0.0f;
  if (k < Kc) {
    int khw = KH * KW;
    int ci = k / khw;
    int r  = k % khw;
    int kh = r / KW;
    int kw = r % KW;
    int wo = (int)(col % Wo);
    long t = col / Wo;
    int ho = (int)(t % Ho);
    int n  = (int)(t / Ho);
    int h = ho * stride - pad + kh;
    int w = wo * stride - pad + kw;
    if (h >= 0 && h < H && w >= 0 && w < W)
      v = src[(size_t)ci * strideC + (size_t)n * strideN + (size_t)h * W + w];
  }
  dst[idx] = (_Float16)v;
}

// ---------------------------------------------------------------------------
// WMMA GEMM with async-LDS double buffering.
//   C[M x Ncols] (f32) = A[Mpad64 x Kpad] (f16 row-major)
//                      * B[Ncols x Kpad]  (f16, each column's K contiguous)
// Block = 128 threads = 4 waves; block tile = 64(M) x 64(N).
// Per k-chunk (K=32): the 64x32 B sub-tile (4 KB) is DMA'd into LDS once via
// global_load_async_to_lds_b128 (double buffered, overlapping WMMA), then all
// 4 waves read their B fragments from LDS. Each wave holds one 16-row A
// fragment (global, L2-resident weights) and 4 f32 accumulators.
// LDS tile: 64 cols x 32 halves, column stride padded 64B->80B to spread
// banks for fragment reads.
// ---------------------------------------------------------------------------
__global__ void __launch_bounds__(128)
k_gemm_wmma(const _Float16* __restrict__ A, const _Float16* __restrict__ B,
            float* __restrict__ C, int M, int Kpad, long Ncols) {
  __shared__ _Float16 sB[2][64 * 40];  // 2 x 2560 halves = 10 KB

  const int t     = threadIdx.x;   // 0..127
  const int wave  = t >> 5;
  const int lane  = t & 31;
  const int row16 = lane & 15;
  const int hi    = lane >> 4;

  const long colBase = (long)blockIdx.x * 64;
  const int  mBase   = blockIdx.y * 64 + wave * 16;

  // A fragment base (contiguous K layout matching B: lane<16 -> K {0..15},
  // lane>=16 -> K {16..31} of row lane&15) => two v8h loads per step.
  const _Float16* aPtr = A + (size_t)(mBase + row16) * Kpad + hi * 16;

  // Staging: 64 cols x 4 chunks of 16B = 256 chunks; thread t copies 2.
  const int q0 = t * 2, q1 = q0 + 1;
  const int col0 = q0 >> 2, c0 = q0 & 3;
  const int col1 = q1 >> 2, c1 = q1 & 3;
  const _Float16* g0 = B + (size_t)(colBase + col0) * Kpad + c0 * 8;
  const _Float16* g1 = B + (size_t)(colBase + col1) * Kpad + c1 * 8;
  const int l0 = col0 * 40 + c0 * 8;
  const int l1 = col1 * 40 + c1 * 8;

  v8f acc0 = {}, acc1 = {}, acc2 = {}, acc3 = {};

  // Prologue: stage k-chunk 0 into buffer 0.
  async_b128_to_lds(g0, &sB[0][l0]);
  async_b128_to_lds(g1, &sB[0][l1]);

  const int nk = Kpad >> 5;
  for (int ks = 0; ks < nk; ++ks) {
    const int cur = ks & 1;
    wait_async0();       // this wave's copies for buffer `cur` have landed
    __syncthreads();     // all waves' copies visible; prev reads of nxt done

    if (ks + 1 < nk) {   // overlap next chunk's DMA with this chunk's WMMAs
      const int nxt = cur ^ 1;
      const int kofs = (ks + 1) * 32;
      async_b128_to_lds(g0 + kofs, &sB[nxt][l0]);
      async_b128_to_lds(g1 + kofs, &sB[nxt][l1]);
    }

    v16h a;
    ((v8h*)&a)[0] = *(const v8h*)(aPtr + ks * 32);
    ((v8h*)&a)[1] = *(const v8h*)(aPtr + ks * 32 + 8);

    // B fragment (32x16): lane holds K hi*16..hi*16+15 of col g*16+row16.
    const _Float16* sbase = &sB[cur][row16 * 40 + hi * 16];
    v16h vb0, vb1, vb2, vb3;
    ((v8h*)&vb0)[0] = *(const v8h*)(sbase + 0 * 640);
    ((v8h*)&vb0)[1] = *(const v8h*)(sbase + 0 * 640 + 8);
    ((v8h*)&vb1)[0] = *(const v8h*)(sbase + 1 * 640);
    ((v8h*)&vb1)[1] = *(const v8h*)(sbase + 1 * 640 + 8);
    ((v8h*)&vb2)[0] = *(const v8h*)(sbase + 2 * 640);
    ((v8h*)&vb2)[1] = *(const v8h*)(sbase + 2 * 640 + 8);
    ((v8h*)&vb3)[0] = *(const v8h*)(sbase + 3 * 640);
    ((v8h*)&vb3)[1] = *(const v8h*)(sbase + 3 * 640 + 8);

    acc0 = __builtin_amdgcn_wmma_f32_16x16x32_f16(false, a, false, vb0, (short)0, acc0, false, false);
    acc1 = __builtin_amdgcn_wmma_f32_16x16x32_f16(false, a, false, vb1, (short)0, acc1, false, false);
    acc2 = __builtin_amdgcn_wmma_f32_16x16x32_f16(false, a, false, vb2, (short)0, acc2, false, false);
    acc3 = __builtin_amdgcn_wmma_f32_16x16x32_f16(false, a, false, vb3, (short)0, acc3, false, false);
  }

  // C layout: lane -> N=row16 (+16/32/48), VGPR r -> M = hi*8 + r.
  for (int r = 0; r < 8; ++r) {
    int row = mBase + hi * 8 + r;
    if (row < M) {
      size_t base = (size_t)row * Ncols + colBase + row16;
      C[base +  0] = acc0[r];
      C[base + 16] = acc1[r];
      C[base + 32] = acc2[r];
      C[base + 48] = acc3[r];
    }
  }
}

// ---------------------------------------------------------------------------
// Per-channel mean/var (channel rows are contiguous). One block per channel.
// ---------------------------------------------------------------------------
__global__ void k_bn_stats(const float* __restrict__ x, float* __restrict__ mv,
                           long Ncols) {
  int c = blockIdx.x;
  const float* row = x + (size_t)c * Ncols;
  float s = 0.f, sq = 0.f;
  for (long i = threadIdx.x; i < Ncols; i += blockDim.x) {
    float v = row[i];
    s += v;
    sq += v * v;
  }
  __shared__ float ssum[256];
  __shared__ float ssq[256];
  ssum[threadIdx.x] = s;
  ssq[threadIdx.x]  = sq;
  __syncthreads();
  for (int o = 128; o > 0; o >>= 1) {
    if ((int)threadIdx.x < o) {
      ssum[threadIdx.x] += ssum[threadIdx.x + o];
      ssq[threadIdx.x]  += ssq[threadIdx.x + o];
    }
    __syncthreads();
  }
  if (threadIdx.x == 0) {
    float m = ssum[0] / (float)Ncols;
    float v = ssq[0] / (float)Ncols - m * m;
    mv[2 * c]     = m;
    mv[2 * c + 1] = v;
  }
}

// ---------------------------------------------------------------------------
// In-place batchnorm + leaky relu, channel-major activations.
// ---------------------------------------------------------------------------
__global__ void k_bn_leaky(float* __restrict__ x, const float* __restrict__ mv,
                           const float* __restrict__ g, const float* __restrict__ b,
                           long Ncols, long total) {
  long idx = (long)blockIdx.x * blockDim.x + threadIdx.x;
  if (idx >= total) return;
  int c = (int)(idx / Ncols);
  float m = mv[2 * c];
  float v = mv[2 * c + 1];
  float y = (x[idx] - m) * rsqrtf(v + BN_EPS) * g[c] + b[c];
  x[idx] = (y >= 0.f) ? y : LSLOPE * y;
}

// ---------------------------------------------------------------------------
// Build f16 B matrix for the final linear: leaky(conv5out), K 1000->1024,
// column-major K-vectors. conv5out is [1000 x 1472] row-major.
// ---------------------------------------------------------------------------
__global__ void k_linb(const float* __restrict__ conv5, _Float16* __restrict__ dst,
                       int Kc, int Kpad, long Ncols, long total) {
  long idx = (long)blockIdx.x * blockDim.x + threadIdx.x;
  if (idx >= total) return;
  int  c   = (int)(idx % Kpad);
  long col = idx / Kpad;
  float v = 0.f;
  if (c < Kc) {
    v = conv5[(size_t)c * Ncols + col];
    v = (v >= 0.f) ? v : LSLOPE * v;
  }
  dst[idx] = (_Float16)v;
}

// ---------------------------------------------------------------------------
// Final: sigmoid + a/bh recombination + transpose.
// lin layout: [64 rows o][1472 cols = b*23 + j]. Output flat b*768 + j2*32 + o.
// ---------------------------------------------------------------------------
__global__ void k_final(const float* __restrict__ lin, float* __restrict__ out) {
  int idx = blockIdx.x * blockDim.x + threadIdx.x;
  if (idx >= 64 * 24 * 32) return;
  int o  = idx & 31;
  int j2 = (idx >> 5) % 24;
  int b  = idx / (24 * 32);
  const long LD = 1472;
  const long cb = (long)b * 23;
  float r;
  if (j2 == 0) {
    float v = lin[(size_t)o * LD + cb + 0];
    r = 1.f / (1.f + __expf(-v));
  } else if (j2 == 23) {
    float v = lin[(size_t)(32 + o) * LD + cb + 22];
    r = 1.f / (1.f + __expf(-v));
  } else {
    float va = lin[(size_t)o * LD + cb + j2];
    float vb = lin[(size_t)(32 + o) * LD + cb + (j2 - 1)];
    float sa = 1.f / (1.f + __expf(-va));
    float sb = 1.f / (1.f + __expf(-vb));
    r = 0.5f * (sa + sb);
  }
  out[idx] = r;
}

// ---------------------------------------------------------------------------
// Host-side orchestration
// ---------------------------------------------------------------------------
struct LayerDesc {
  int Ci, H, W, Ho, Wo, KH, KW, stride, pad;
  int Co, Kc, Kpad, Mpad64;
  long strideC, strideN, Ncols;
};

extern "C" void kernel_launch(void* const* d_in, const int* in_sizes, int n_in,
                              void* d_out, int out_size, void* d_ws, size_t ws_size,
                              hipStream_t stream) {
  (void)in_sizes; (void)n_in; (void)out_size; (void)ws_size;

  const float* x = (const float*)d_in[0];
  const float* Wt[6] = {(const float*)d_in[1],  (const float*)d_in[4],
                        (const float*)d_in[7],  (const float*)d_in[10],
                        (const float*)d_in[13], (const float*)d_in[16]};
  const float* Gm[5] = {(const float*)d_in[2], (const float*)d_in[5],
                        (const float*)d_in[8], (const float*)d_in[11],
                        (const float*)d_in[14]};
  const float* Bt[5] = {(const float*)d_in[3], (const float*)d_in[6],
                        (const float*)d_in[9], (const float*)d_in[12],
                        (const float*)d_in[15]};
  const float* linw = (const float*)d_in[17];

  static const LayerDesc L[6] = {
    //  Ci   H    W   Ho  Wo KH KW st pd   Co   Kc  Kpad Mpad64 strideC strideN  Ncols
    {   3, 64, 768, 32, 384, 4, 4, 2, 1,  16,  48,   64,   64,  49152, 147456, 786432},
    {  16, 32, 384, 16, 192, 4, 4, 2, 1,  32, 256,  256,   64, 786432,  12288, 196608},
    {  32, 16, 192,  8,  96, 4, 4, 2, 1,  64, 512,  512,   64, 196608,   3072,  49152},
    {  64,  8,  96,  4,  48, 4, 4, 2, 1, 128,1024, 1024,  128,  49152,    768,  12288},
    { 128,  4,  48,  2,  24, 4, 4, 2, 1, 256,2048, 2048,  256,  12288,    192,   3072},
    { 256,  2,  24,  1,  23, 2, 2, 1, 0,1000,1024, 1024, 1024,   3072,     48,   1472},
  };

  char* ws = (char*)d_ws;
  size_t off = 0;
  auto alloc = [&](size_t bytes) -> char* {
    char* p = ws + off;
    off = (off + bytes + 255) & ~(size_t)255;
    return p;
  };

  // Activations (channel-major [Co][Ncols]).
  float* act[6];
  for (int i = 0; i < 6; ++i)
    act[i] = (float*)alloc((size_t)L[i].Co * L[i].Ncols * sizeof(float));

  // Shared im2col buffer (f16), sized for the largest layers (50,331,648 halves).
  _Float16* colbuf = (_Float16*)alloc((size_t)50331648 * sizeof(_Float16));

  // f16 weights (zero padded to Mpad64 x Kpad).
  _Float16* wp[6];
  for (int i = 0; i < 6; ++i)
    wp[i] = (_Float16*)alloc((size_t)L[i].Mpad64 * L[i].Kpad * sizeof(_Float16));
  _Float16* wlin = (_Float16*)alloc((size_t)64 * 1024 * sizeof(_Float16));

  // BN stats, reused per layer.
  float* mv = (float*)alloc(2 * 256 * sizeof(float));

  // Linear output [64 x 1472].
  float* lout = (float*)alloc((size_t)64 * 1472 * sizeof(float));

  // --- weight conversion ---
  for (int i = 0; i < 6; ++i) {
    int tot = L[i].Mpad64 * L[i].Kpad;
    k_wcvt<<<(tot + 255) / 256, 256, 0, stream>>>(Wt[i], wp[i], L[i].Co, L[i].Kc,
                                                  L[i].Kpad, L[i].Mpad64);
  }
  k_wcvt<<<(64 * 1024 + 255) / 256, 256, 0, stream>>>(linw, wlin, 64, 1000, 1024, 64);

  // --- conv layers ---
  const float* src = x;
  for (int i = 0; i < 6; ++i) {
    const LayerDesc& d = L[i];
    long totalCol = d.Ncols * (long)d.Kpad;
    k_im2col<<<(unsigned)((totalCol + 255) / 256), 256, 0, stream>>>(
        src, colbuf, d.H, d.W, d.Ho, d.Wo, d.KH, d.KW, d.stride, d.pad,
        d.strideC, d.strideN, d.Kc, d.Kpad, totalCol);

    dim3 grid((unsigned)(d.Ncols / 64), (unsigned)(d.Mpad64 / 64));
    k_gemm_wmma<<<grid, 128, 0, stream>>>(wp[i], colbuf, act[i], d.Co, d.Kpad, d.Ncols);

    if (i < 5) {
      k_bn_stats<<<d.Co, 256, 0, stream>>>(act[i], mv, d.Ncols);
      long tot = (long)d.Co * d.Ncols;
      k_bn_leaky<<<(unsigned)((tot + 255) / 256), 256, 0, stream>>>(
          act[i], mv, Gm[i], Bt[i], d.Ncols, tot);
    }
    src = act[i];
  }

  // --- leaky + linear (einsum) via WMMA GEMM: [64 x 1024] * [1024 x 1472] ---
  {
    long totalB = (long)1472 * 1024;
    k_linb<<<(unsigned)((totalB + 255) / 256), 256, 0, stream>>>(
        act[5], colbuf, 1000, 1024, 1472, totalB);
    dim3 grid(1472 / 64, 1);
    k_gemm_wmma<<<grid, 128, 0, stream>>>(wlin, colbuf, lout, 64, 1024, 1472);
  }

  // --- sigmoid + recombination + transpose ---
  k_final<<<(64 * 24 * 32 + 255) / 256, 256, 0, stream>>>(lout, (float*)d_out);
}
